// Layer3bitWithZero_83356725281469
// MI455X (gfx1250) — matrix operand order repaired and checked by reading, not compiled, not run on hardware
//
#include <hip/hip_runtime.h>
#include <stdint.h>

typedef __attribute__((ext_vector_type(16))) _Float16 v16h;
typedef __attribute__((ext_vector_type(8)))  _Float16 v8h;
typedef __attribute__((ext_vector_type(4)))  _Float16 v4h;
typedef __attribute__((ext_vector_type(8)))  float    v8f;
typedef __attribute__((ext_vector_type(4)))  int      v4i;

#define BM 128
#define BN 128
#define BK 64
#define LDP 72   // padded LDS stride in halves (144 B -> 16B-aligned rows)

// CDNA5 async global->LDS copy (per-lane 16 bytes), tracked by ASYNCcnt.
__device__ __forceinline__ void async_ld16(uint32_t lds_byte, const void* g) {
  asm volatile("global_load_async_to_lds_b128 %0, %1, off"
               :: "v"(lds_byte), "v"(g) : "memory");
}
__device__ __forceinline__ void wait_async0() {
  asm volatile("s_wait_asynccnt 0x0" ::: "memory");
}

__global__ __launch_bounds__(256)
void w3a16_wmma_gemm(const _Float16* __restrict__ A,
                     const int* __restrict__ q,
                     const _Float16* __restrict__ s,
                     const _Float16* __restrict__ z,
                     _Float16* __restrict__ out,
                     int M, int K, int N) {
  __shared__ _Float16 As[2][BM][LDP];   // [buf][m][k]
  __shared__ _Float16 Ws[2][BN][LDP];   // [buf][n][k] transposed, dequantized

  const int tid  = threadIdx.x;
  const int lane = tid & 31;
  const int wave = tid >> 5;
  const int lh   = lane >> 4;
  const int l16  = lane & 15;

  const int bm = blockIdx.y * BM;
  const int bn = blockIdx.x * BN;

  const int wm = (wave & 1) * 64;   // 2 waves along M
  const int wn = (wave >> 1) * 32;  // 4 waves along N

  v8f acc[4][2] = {};

  // A tile mapping: 8 threads/row * 8 halves, 32 rows/pass, 4 passes
  const int arow = tid >> 3;
  const int acol = (tid & 7) * 8;

  // q mapping: 4 adjacent columns per thread, 8 strided k-rows
  const int n4   = (tid & 31) * 4;   // column group: 0,4,...,124
  const int krow = tid >> 5;         // base k row: 0..7 (step 8 below)

  v4i  qreg[8];
  v4h  sv4  = {};
  v4h  zadj = {};                    // z - 8*s  (folded zero-point)

  auto loadA = [&](int buf, int k0) {
    #pragma unroll
    for (int r = 0; r < 4; ++r) {
      const int row = arow + r * 32;
      const uint32_t lds = (uint32_t)(uintptr_t)&As[buf][row][acol];
      async_ld16(lds, (const void*)(A + (size_t)(bm + row) * K + k0 + acol));
    }
  };

  auto loadQ = [&](int k0) {
    const int g = k0 >> 6;  // BK == GROUPSIZE
    const _Float16* sp = s + (size_t)g * N + bn + n4;
    const _Float16* zp = z + (size_t)g * N + bn + n4;
    const v4h sraw = *(const v4h*)sp;       // 8B load, aligned
    const v4h zraw = *(const v4h*)zp;
    #pragma unroll
    for (int c = 0; c < 4; ++c) {
      sv4[c]  = sraw[c];
      zadj[c] = zraw[c] - (_Float16)8.0f * sraw[c];
    }
    // uniform base + 32-bit thread offset -> saddr-form global_load_b128
    const int* qb = q + (size_t)k0 * N + bn;
    const uint32_t ob = (uint32_t)(krow * N + n4);
    #pragma unroll
    for (int p = 0; p < 8; ++p)
      qreg[p] = *(const v4i*)(qb + (ob + (uint32_t)(8 * p) * (uint32_t)N));
  };

  auto dequant = [&](int buf) {
    #pragma unroll
    for (int p = 0; p < 8; ++p) {
      const int k = krow + 8 * p;
      #pragma unroll
      for (int c = 0; c < 4; ++c) {
        // fp16 bits of (8+q): 0x4800 | (q<<7); then (8+q)*s + (z-8s) = q*s+z
        union { unsigned short u; _Float16 h; } cv;
        cv.u = (unsigned short)(0x4800u | ((unsigned)qreg[p][c] << 7));
        Ws[buf][n4 + c][k] = cv.h * sv4[c] + zadj[c];   // v_fma_f16
      }
    }
  };

  auto compute = [&](int buf) {
    #pragma unroll
    for (int ks = 0; ks < 2; ++ks) {
      const int kk = ks * 32;
      v16h bfrag[2], afrag[4];
      #pragma unroll
      for (int j = 0; j < 2; ++j) {
        const _Float16* bp = &Ws[buf][wn + j * 16 + l16][kk + lh * 8];
        const v8h lo = *(const v8h*)bp;
        const v8h hi = *(const v8h*)(bp + 16);
        #pragma unroll
        for (int e = 0; e < 8; ++e) { bfrag[j][e] = lo[e]; bfrag[j][8 + e] = hi[e]; }
      }
      #pragma unroll
      for (int i = 0; i < 4; ++i) {
        const _Float16* ap = &As[buf][wm + i * 16 + l16][kk + lh * 8];
        const v8h lo = *(const v8h*)ap;
        const v8h hi = *(const v8h*)(ap + 16);
        #pragma unroll
        for (int e = 0; e < 8; ++e) { afrag[i][e] = lo[e]; afrag[i][8 + e] = hi[e]; }
      }
      #pragma unroll
      for (int i = 0; i < 4; ++i)
        #pragma unroll
        for (int j = 0; j < 2; ++j)
          acc[i][j] = __builtin_amdgcn_wmma_f32_16x16x32_f16(
              false, afrag[i], false, bfrag[j], (short)0, acc[i][j], false, false);
    }
  };

  // ---- pipeline prologue: stage tile 0 ----
  loadA(0, 0);
  loadQ(0);
  dequant(0);
  wait_async0();
  __syncthreads();

  int buf = 0;
  for (int k0 = 0; k0 < K; k0 += BK) {
    const int nxt = k0 + BK;
    const bool more = nxt < K;

    if (more) {
      loadA(buf ^ 1, nxt);   // async: fills next A buffer while we compute
      loadQ(nxt);            // b128 q loads into VGPRs, hidden by WMMAs
    }

    compute(buf);

    if (more) {
      dequant(buf ^ 1);      // bit-trick dequant -> next W buffer
      wait_async0();         // next A buffer resident before the barrier
    }
    __syncthreads();
    buf ^= 1;
  }

  // ---- epilogue: f32 -> f16; C/D layout: n = l16, M = lh*8 + r ----
  #pragma unroll
  for (int i = 0; i < 4; ++i) {
    #pragma unroll
    for (int j = 0; j < 2; ++j) {
      const int n = bn + wn + j * 16 + l16;
      #pragma unroll
      for (int r = 0; r < 8; ++r) {
        const int m = bm + wm + i * 16 + lh * 8 + r;
        out[(size_t)m * N + n] = (_Float16)acc[i][j][r];
      }
    }
  }
}

extern "C" void kernel_launch(void* const* d_in, const int* in_sizes, int n_in,
                              void* d_out, int out_size, void* d_ws, size_t ws_size,
                              hipStream_t stream) {
  const _Float16* A = (const _Float16*)d_in[0];
  const int*      q = (const int*)d_in[1];
  const _Float16* s = (const _Float16*)d_in[2];
  const _Float16* z = (const _Float16*)d_in[3];
  _Float16*     out = (_Float16*)d_out;

  const long long sizeA = in_sizes[0];   // M*K
  const long long sizeQ = in_sizes[1];   // K*N
  const long long sizeS = in_sizes[2];   // (K/64)*N

  const int K = (int)(64LL * sizeQ / sizeS);
  const int N = (int)(sizeQ / K);
  const int M = (int)(sizeA / K);

  dim3 grid((unsigned)(N / BN), (unsigned)(M / BM));
  dim3 block(256);
  w3a16_wmma_gemm<<<grid, block, 0, stream>>>(A, q, s, z, out, M, K, N);
}
